// IDECClusteringHead_43688407335119
// MI455X (gfx1250) — matrix operand order repaired and checked by reading, not compiled, not run on hardware
//
#include <hip/hip_runtime.h>
#include <hip/hip_bf16.h>

// ---------------------------------------------------------------------------
// IDEC clustering head for MI455X (gfx1250, wave32).
//   dist[n,k] = ||z_n||^2 + ||c_k||^2 - 2 z_n . c_k    (cross term via bf16 WMMA)
//   q = 1/(1+dist);  q /= (rowsum(q) + 1e-8)
// N=131072, K=256, D=256.  HBM floor ~268MB => bf16 WMMA keeps us memory-bound.
// z streamed once (NT loads), q written once (NT stores); centers stay in L2/LDS.
// ---------------------------------------------------------------------------

typedef __attribute__((ext_vector_type(16))) __bf16 v16bf;
typedef __attribute__((ext_vector_type(2)))  __bf16 v2bf;
typedef __attribute__((ext_vector_type(8)))  float  v8f;
typedef __attribute__((ext_vector_type(4)))  float  v4f;
typedef __attribute__((ext_vector_type(2)))  float  v2f;

#define D_DIM 256
#define K_DIM 256
#define ROWS_PER_WG 128      // 8 waves x 16 rows
#define C_ROWS 128           // center rows staged in LDS per pass (2 passes)
#define C_STRIDE 264         // 256 + 8 bf16 pad: row stride = 132 dwords == 4 mod 64
                             // -> ds_load_b128 across 16 lanes tiles all 64 banks

// two floats -> packed 2x bf16 via IR fptrunc; backend picks v_cvt_pk_bf16_f32
// on gfx1250 (bf16-cvt feature), software RNE otherwise.
static __device__ inline v2bf cvt2bf(float a, float b) {
    v2f t; t[0] = a; t[1] = b;
    return __builtin_convertvector(t, v2bf);
}

__global__ __launch_bounds__(256, 1)
void idec_head_kernel(const float* __restrict__ z,
                      const float* __restrict__ centers,
                      float* __restrict__ out, int N) {
    __shared__ alignas(16) __bf16 sC[C_ROWS][C_STRIDE];   // bf16 centers tile (~66KB)
    __shared__ float s_csq[K_DIM][2];                     // fp32 ||c_k||^2 partials

    const int tid  = threadIdx.x;
    const int lane = tid & 31;
    const int wv   = tid >> 5;          // wave 0..7
    const int half = lane >> 4;         // 0|1 (wave half)
    const int l16  = lane & 15;

    const int waveRow = blockIdx.x * ROWS_PER_WG + wv * 16;

    // ---- Preload A: 16 rows x 256 d as bf16 (ISA 16-bit A 16x32 layout:
    // lane holds row M=l16; per 32-chunk, K = 8*half + {0..7} and 16+8*half + {0..7}).
    // Also accumulate fp32 ||z||^2 from the original data. z is streamed once -> NT.
    v16bf Areg[8];
    float zp = 0.f;
    const v4f* zrp = (const v4f*)(z + (size_t)(waveRow + l16) * D_DIM + (half << 3));
    #pragma unroll
    for (int kc = 0; kc < 8; ++kc) {
        const v4f* p = zrp + kc * 8;           // 32 floats per chunk
        v4f f0 = __builtin_nontemporal_load(p + 0);
        v4f f1 = __builtin_nontemporal_load(p + 1);
        v4f f2 = __builtin_nontemporal_load(p + 4);
        v4f f3 = __builtin_nontemporal_load(p + 5);
        #pragma unroll
        for (int e = 0; e < 4; ++e) {
            zp = fmaf(f0[e], f0[e], zp); zp = fmaf(f1[e], f1[e], zp);
            zp = fmaf(f2[e], f2[e], zp); zp = fmaf(f3[e], f3[e], zp);
        }
        union { v16bf v; v2bf h[8]; } a;
        a.h[0] = cvt2bf(f0[0], f0[1]);  a.h[1] = cvt2bf(f0[2], f0[3]);
        a.h[2] = cvt2bf(f1[0], f1[1]);  a.h[3] = cvt2bf(f1[2], f1[3]);
        a.h[4] = cvt2bf(f2[0], f2[1]);  a.h[5] = cvt2bf(f2[2], f2[3]);
        a.h[6] = cvt2bf(f3[0], f3[1]);  a.h[7] = cvt2bf(f3[2], f3[3]);
        Areg[kc] = a.v;
    }
    // combine K-halves: lane L now holds ||z||^2 of row waveRow + (L&15)
    float zsq = zp + __shfl_xor(zp, 16, 32);

    v8f acc[16];
    #pragma unroll
    for (int i = 0; i < 16; ++i)
        acc[i] = (v8f){0.f,0.f,0.f,0.f,0.f,0.f,0.f,0.f};

    // ---- Two passes over the 256 clusters (128 center rows in LDS each) ----
    #pragma unroll
    for (int pass = 0; pass < 2; ++pass) {
        __syncthreads();   // pass 1: all waves done reading pass-0 sC
        {   // stage centers -> LDS bf16; fp32 column norms. 2 threads per row.
            const int r  = tid >> 1;
            const int h2 = tid & 1;
            const float* src = centers + (size_t)(pass * C_ROWS + r) * D_DIM + h2 * 128;
            __bf16* dst = &sC[r][h2 * 128];
            float sq = 0.f;
            #pragma unroll 4
            for (int i = 0; i < 128; i += 4) {
                v4f f = *(const v4f*)(src + i);   // centers: keep RT (L2-resident)
                sq = fmaf(f[0],f[0],sq); sq = fmaf(f[1],f[1],sq);
                sq = fmaf(f[2],f[2],sq); sq = fmaf(f[3],f[3],sq);
                union { v2bf h[2]; uint2 u; } pkd;
                pkd.h[0] = cvt2bf(f[0], f[1]);
                pkd.h[1] = cvt2bf(f[2], f[3]);
                *(uint2*)(dst + i) = pkd.u;
            }
            s_csq[pass * C_ROWS + r][h2] = sq;
        }
        __syncthreads();

        // B layout (16-bit B 32x16): lane = column n; per half, K = 8*half + {0..7}
        // and 16+8*half + {0..7} (same K order as A).
        const int kb_half = half << 3;
        #pragma unroll
        for (int kc = 0; kc < 8; ++kc) {
            v16bf a = Areg[kc];
            #pragma unroll
            for (int t = 0; t < 8; ++t) {
                union { v16bf v; uint4 q[2]; } b;
                const uint4* bp = (const uint4*)&sC[t * 16 + l16][kc * 32 + kb_half];
                b.q[0] = bp[0];
                b.q[1] = bp[2];
                acc[pass * 8 + t] = __builtin_amdgcn_wmma_f32_16x16x32_bf16(
                    false, a, false, b.v, (short)0, acc[pass * 8 + t], false, false);
            }
        }
    }

    // ---- Epilogue: dist -> q -> row-normalize -> store ----
    float csq_t[16];
    #pragma unroll
    for (int t = 0; t < 16; ++t) {
        int c = t * 16 + l16;
        csq_t[t] = s_csq[c][0] + s_csq[c][1];
    }

    // C layout: lane = n (l16), VGPR j = row m = 8*half + j
    #pragma unroll
    for (int j = 0; j < 8; ++j) {
        const int m = (half << 3) + j;
        float zsq_m = __shfl(zsq, m, 32);          // row norm for this VGPR's row
        float qv[16];
        float rs = 0.f;
        #pragma unroll
        for (int t = 0; t < 16; ++t) {
            float d = zsq_m + csq_t[t] - 2.f * acc[t][j];
            float q = __builtin_amdgcn_rcpf(1.f + d);
            qv[t] = q;
            rs += q;
        }
        // row sum across the 16 lanes of this half-wave
        rs += __shfl_xor(rs, 1, 32);
        rs += __shfl_xor(rs, 2, 32);
        rs += __shfl_xor(rs, 4, 32);
        rs += __shfl_xor(rs, 8, 32);
        float inv = __builtin_amdgcn_rcpf(rs + 1e-8f);
        float* op = out + (size_t)(waveRow + m) * K_DIM + l16;
        #pragma unroll
        for (int t = 0; t < 16; ++t)
            __builtin_nontemporal_store(qv[t] * inv, op + t * 16);  // 64B/half-wave
    }
    (void)N;
}

extern "C" void kernel_launch(void* const* d_in, const int* in_sizes, int n_in,
                              void* d_out, int out_size, void* d_ws, size_t ws_size,
                              hipStream_t stream) {
    const float* z       = (const float*)d_in[0];
    const float* centers = (const float*)d_in[1];
    float* out           = (float*)d_out;
    const int N = in_sizes[0] / D_DIM;          // 131072
    const int grid = N / ROWS_PER_WG;           // 1024 workgroups
    idec_head_kernel<<<dim3(grid), dim3(256), 0, stream>>>(z, centers, out, N);
    (void)n_in; (void)out_size; (void)d_ws; (void)ws_size;
}
